// MultiHeadSelfAttention_1125281432241
// MI455X (gfx1250) — matrix operand order repaired and checked by reading, not compiled, not run on hardware
//
#include <hip/hip_runtime.h>
#include <hip/hip_bf16.h>

// ---------------------------------------------------------------------------
// MHSA forward for MI455X (gfx1250): bf16 WMMA GEMMs + flash attention,
// async global->LDS double-buffered staging, global prefetch.
// B=4, S=2048, D=1024, H=16, HD=64.
// ---------------------------------------------------------------------------

typedef __attribute__((ext_vector_type(16))) __bf16 v16bf;
typedef __attribute__((ext_vector_type(8)))  __bf16 v8bf;
typedef __attribute__((ext_vector_type(4)))  __bf16 v4bf;
typedef __attribute__((ext_vector_type(8)))  float  v8f;

#define WMMA_BF16(a, b, c) \
  __builtin_amdgcn_wmma_f32_16x16x32_bf16(false, (a), false, (b), (short)0, (c), false, false)

#if __has_builtin(__builtin_amdgcn_global_load_async_to_lds_b128)
#define HAVE_ASYNC_LDS 1
// Exact parameter types per hipcc diagnostic: pointer to 16B int vector,
// global (AS1) source and LDS (AS3) destination.
typedef int v4i_16 __attribute__((vector_size(16)));
typedef __attribute__((address_space(1))) v4i_16* as1_v4i_ptr;
typedef __attribute__((address_space(3))) v4i_16* as3_v4i_ptr;
#else
#define HAVE_ASYNC_LDS 0
#endif

// Move 16 bytes global -> LDS. Async (ASYNCcnt-tracked, no VGPR data) when the
// toolchain exposes the gfx1250 builtin; plain load+ds_store otherwise.
__device__ __forceinline__ void stage16(const __bf16* g, __bf16* l) {
#if HAVE_ASYNC_LDS
  __builtin_amdgcn_global_load_async_to_lds_b128((as1_v4i_ptr)g, (as3_v4i_ptr)l,
                                                 0, 0);
#else
  *(v8bf*)l = *(const v8bf*)g;
#endif
}

__device__ __forceinline__ void wait_async_lds() {
#if HAVE_ASYNC_LDS
  asm volatile("s_wait_asynccnt 0" ::: "memory");
#endif
}

// A-matrix 16x32 bf16 fragment: lane = h*16 + m (m = lane%16).
// Per lane: elements 0..15 -> K = 16h + e (contiguous, matches B fragments)
__device__ __forceinline__ v16bf load_afrag(const __bf16* rowptr, int h) {
  v8bf lo = *(const v8bf*)(rowptr + 16 * h);
  v8bf hi = *(const v8bf*)(rowptr + 16 * h + 8);
  return __builtin_shufflevector(lo, hi, 0, 1, 2, 3, 4, 5, 6, 7,
                                 8, 9, 10, 11, 12, 13, 14, 15);
}

// ---------------------------------------------------------------------------
// Stage 0: f32 -> bf16 conversion, 4-wide (pure bandwidth)
// ---------------------------------------------------------------------------
__global__ void cvt_f32_bf16(const float4* __restrict__ src,
                             __bf16* __restrict__ dst, int n4) {
  int i = blockIdx.x * blockDim.x + threadIdx.x;
  if (i < n4) {
    float4 f = src[i];
    v4bf o = {(__bf16)f.x, (__bf16)f.y, (__bf16)f.z, (__bf16)f.w};
    *(v4bf*)(dst + 4 * i) = o;
  }
}

// ---------------------------------------------------------------------------
// Shared GEMM core: 4 waves, 64x64 C tile, K=1024, A staged through
// double-buffered LDS with async copies; B rows read straight from global.
// Wave w owns N-subtile 16w and all four 16-row M-subtiles.
// ---------------------------------------------------------------------------
#define GEMM_CORE(Aptr, Bptr)                                                  \
  __shared__ __bf16 lsA[2][64 * 32];                                           \
  const int tid  = threadIdx.x;                                                \
  const int wave = tid >> 5;                                                   \
  const int lane = tid & 31;                                                   \
  const int h    = lane >> 4;                                                  \
  const int l16  = lane & 15;                                                  \
  const int m0   = blockIdx.y * 64;                                            \
  const int n0   = blockIdx.x * 64 + wave * 16;                                \
  const int ldrow  = tid >> 1;                                                 \
  const int ldhalf = tid & 1;                                                  \
  const __bf16* gstage = Aptr + (size_t)(m0 + ldrow) * 1024 + ldhalf * 16;     \
  const __bf16* wrow   = Bptr + (size_t)(n0 + l16) * 1024 + h * 16;            \
  __bf16* lstage = &lsA[0][0] + ldrow * 32 + ldhalf * 16;                      \
  v8f acc[4] = {};                                                             \
  /* prologue: stage k0 = 0 into buffer 0 */                                   \
  stage16(gstage, lstage);                                                     \
  stage16(gstage + 8, lstage + 8);                                             \
  wait_async_lds();                                                            \
  __syncthreads();                                                             \
  int buf = 0;                                                                 \
  for (int k0 = 0; k0 < 1024; k0 += 32) {                                      \
    if (k0 + 32 < 1024) { /* stage next tile into other buffer */              \
      __bf16* ln = &lsA[0][0] + (buf ^ 1) * (64 * 32) + ldrow * 32 +           \
                   ldhalf * 16;                                                \
      stage16(gstage + k0 + 32, ln);                                           \
      stage16(gstage + k0 + 40, ln + 8);                                       \
      __builtin_prefetch(wrow + k0 + 64, 0, 0); /* global_prefetch */          \
    }                                                                          \
    v16bf bfrag = *(const v16bf*)(wrow + k0);                                  \
    const __bf16* lbase = &lsA[0][0] + buf * (64 * 32);                        \
    _Pragma("unroll") for (int r = 0; r < 4; ++r) {                            \
      v16bf af = load_afrag(lbase + (r * 16 + l16) * 32, h);                   \
      acc[r] = WMMA_BF16(af, bfrag, acc[r]);                                   \
    }                                                                          \
    wait_async_lds();                                                          \
    __syncthreads();                                                           \
    buf ^= 1;                                                                  \
  }

// ---------------------------------------------------------------------------
// Stage 1: QKV projection. C[8192,3072] = X[8192,1024] * Wqkv[3072,1024]^T
// Epilogue scatters into Q (pre-scaled by 1/8), K, and V-transposed buffers.
// ---------------------------------------------------------------------------
__global__ void __launch_bounds__(128) gemm_qkv(
    const __bf16* __restrict__ Xb,   // [8192,1024]
    const __bf16* __restrict__ Wb,   // [3072,1024]
    __bf16* __restrict__ Qb,         // [B,H,S,64]  (x 0.125)
    __bf16* __restrict__ Kb,         // [B,H,S,64]
    __bf16* __restrict__ Vtb) {      // [B,H,64,S]
  GEMM_CORE(Xb, Wb)

  // C lane layout: M = v + 8h (within 16-tile r), N = l16.
  const int n    = n0 + l16;
  const int w    = n >> 10;    // 0=Q 1=K 2=V
  const int rem  = n & 1023;
  const int head = rem >> 6;
  const int hd   = rem & 63;
#pragma unroll
  for (int r = 0; r < 4; ++r) {
#pragma unroll
    for (int v = 0; v < 8; ++v) {
      int m = m0 + r * 16 + v + h * 8;
      int b = m >> 11;
      int s = m & 2047;
      float val = acc[r][v];
      if (w == 0) {
        Qb[(((size_t)(b * 16 + head) * 2048) + s) * 64 + hd] =
            (__bf16)(val * 0.125f);  // fold 1/sqrt(64) into Q
      } else if (w == 1) {
        Kb[(((size_t)(b * 16 + head) * 2048) + s) * 64 + hd] = (__bf16)val;
      } else {
        Vtb[(((size_t)(b * 16 + head) * 64) + hd) * 2048 + s] = (__bf16)val;
      }
    }
  }
}

// ---------------------------------------------------------------------------
// Stage 3: output projection. Out[8192,1024] = AO*Wo^T + bias (f32 out)
// ---------------------------------------------------------------------------
__global__ void __launch_bounds__(128) gemm_out(
    const __bf16* __restrict__ Ab,   // [8192,1024]
    const __bf16* __restrict__ Wob,  // [1024,1024]
    const float* __restrict__ bias,  // [1024]
    float* __restrict__ Out) {       // [8192,1024]
  GEMM_CORE(Ab, Wob)

  const float bv = bias[n0 + l16];
#pragma unroll
  for (int r = 0; r < 4; ++r)
#pragma unroll
    for (int v = 0; v < 8; ++v)
      Out[(size_t)(m0 + r * 16 + v + h * 8) * 1024 + n0 + l16] =
          acc[r][v] + bv;
}

// ---------------------------------------------------------------------------
// Stage 2: flash attention. One wave = one 16-row q tile; 4 waves per block.
// QK^T (2x2 WMMA), online softmax (shfl_xor row reductions), P relaid via
// per-wave LDS into A layout (s_wait_dscnt fence), PV (4 WMMA) into f32 accs.
// ---------------------------------------------------------------------------
__global__ void __launch_bounds__(128) flash_attn(
    const __bf16* __restrict__ Qb,   // [B,H,S,64], pre-scaled
    const __bf16* __restrict__ Kb,   // [B,H,S,64]
    const __bf16* __restrict__ Vtb,  // [B,H,64,S]
    __bf16* __restrict__ AOb) {      // [B,S,H*64] = [8192,1024]
  __shared__ __bf16 lsP[4][16 * 32];  // per-wave P staging
  const int tid  = threadIdx.x;
  const int wave = tid >> 5;
  const int lane = tid & 31;
  const int h    = lane >> 4;
  const int l16  = lane & 15;
  const int bh   = blockIdx.y;                 // 0..63
  const int q0   = blockIdx.x * 64 + wave * 16;

  const __bf16* Qhead = Qb + (size_t)bh * 2048 * 64;
  const __bf16* Khead = Kb + (size_t)bh * 2048 * 64;
  const __bf16* Vhead = Vtb + (size_t)bh * 64 * 2048;

  // Q fragments (16x64 = two K=32 chunks), resident for the whole kv loop
  v16bf qa[2];
#pragma unroll
  for (int c = 0; c < 2; ++c)
    qa[c] = load_afrag(Qhead + (size_t)(q0 + l16) * 64 + 32 * c, h);

  v8f o[4] = {};
  float Mrow[8], Lrow[8];
#pragma unroll
  for (int v = 0; v < 8; ++v) { Mrow[v] = -3.0e38f; Lrow[v] = 0.0f; }

  __bf16* pws = &lsP[wave][0];
  const int qmax = q0 + 15;

  for (int kv0 = 0; kv0 <= qmax; kv0 += 32) {
    // prefetch next chunk's K rows / V run (speculative, global_prefetch_b8)
    __builtin_prefetch(Khead + (size_t)(kv0 + 32 + l16) * 64, 0, 0);
    __builtin_prefetch(Vhead + (size_t)l16 * 2048 + kv0 + 32, 0, 0);

    // ---- scores: two 16x16 tiles over 32 kv columns -----------------------
    v8f sc[2];
#pragma unroll
    for (int u = 0; u < 2; ++u) {
      const __bf16* krow = Khead + (size_t)(kv0 + u * 16 + l16) * 64;
      v16bf bk0 = *(const v16bf*)(krow + h * 16);        // hd 0..31
      v16bf bk1 = *(const v16bf*)(krow + 32 + h * 16);   // hd 32..63
      v8f t = {};
      t     = WMMA_BF16(qa[0], bk0, t);
      sc[u] = WMMA_BF16(qa[1], bk1, t);
    }
    // ---- causal mask ------------------------------------------------------
    if (kv0 + 31 > q0) {
#pragma unroll
      for (int u = 0; u < 2; ++u) {
        int n = kv0 + u * 16 + l16;
#pragma unroll
        for (int v = 0; v < 8; ++v)
          if (n > q0 + v + h * 8) sc[u][v] = -3.0e38f;
      }
    }
    // ---- online softmax (row v+8h lives across a 16-lane group) -----------
    float alpha[8];
#pragma unroll
    for (int v = 0; v < 8; ++v) {
      float rm = fmaxf(sc[0][v], sc[1][v]);
#pragma unroll
      for (int off = 8; off >= 1; off >>= 1)
        rm = fmaxf(rm, __shfl_xor(rm, off, 32));
      float mnew = fmaxf(Mrow[v], rm);
      alpha[v]   = __expf(Mrow[v] - mnew);
      Mrow[v]    = mnew;
      float p0   = __expf(sc[0][v] - mnew);
      float p1   = __expf(sc[1][v] - mnew);
      sc[0][v] = p0;
      sc[1][v] = p1;
      float rs = p0 + p1;
#pragma unroll
      for (int off = 8; off >= 1; off >>= 1)
        rs += __shfl_xor(rs, off, 32);
      Lrow[v] = Lrow[v] * alpha[v] + rs;
    }
#pragma unroll
    for (int t = 0; t < 4; ++t)
#pragma unroll
      for (int v = 0; v < 8; ++v) o[t][v] *= alpha[v];

    // ---- P: C layout -> LDS -> A layout fragment --------------------------
#pragma unroll
    for (int u = 0; u < 2; ++u)
#pragma unroll
      for (int v = 0; v < 8; ++v)
        pws[(v + h * 8) * 32 + u * 16 + l16] = (__bf16)sc[u][v];
    asm volatile("s_wait_dscnt 0" ::: "memory");  // intra-wave DS RAW fence
    v16bf pa = load_afrag(pws + l16 * 32, h);

    // ---- PV: B frag from V^T is contiguous along s ------------------------
#pragma unroll
    for (int t = 0; t < 4; ++t) {
      v16bf bv =
          *(const v16bf*)(Vhead + (size_t)(t * 16 + l16) * 2048 + kv0 + h * 16);
      o[t] = WMMA_BF16(pa, bv, o[t]);
    }
    asm volatile("" ::: "memory");
  }

  // ---- normalize + store [B,S,H,64] ---------------------------------------
  const int b = bh >> 4, head = bh & 15;
#pragma unroll
  for (int v = 0; v < 8; ++v) {
    float invl = 1.0f / Lrow[v];
    int srow   = q0 + v + h * 8;
    size_t base = ((size_t)b * 2048 + srow) * 1024 + head * 64;
#pragma unroll
    for (int t = 0; t < 4; ++t)
      AOb[base + t * 16 + l16] = (__bf16)(o[t][v] * invl);
  }
}

// ---------------------------------------------------------------------------
// Host launcher
// ---------------------------------------------------------------------------
extern "C" void kernel_launch(void* const* d_in, const int* in_sizes, int n_in,
                              void* d_out, int out_size, void* d_ws,
                              size_t ws_size, hipStream_t stream) {
  const float* x    = (const float*)d_in[0];  // [4,2048,1024]
  const float* Wqkv = (const float*)d_in[1];  // [3072,1024]
  const float* Wo_w = (const float*)d_in[2];  // [1024,1024]
  const float* Wo_b = (const float*)d_in[3];  // [1024]
  float* out = (float*)d_out;                 // [4,2048,1024]

  char* ws = (char*)d_ws;
  const size_t MB = 1024u * 1024u;
  __bf16* xb  = (__bf16*)(ws);             // 16 MB  x bf16
  __bf16* wqb = (__bf16*)(ws + 16 * MB);   //  6 MB  Wqkv bf16
  __bf16* wob = (__bf16*)(ws + 22 * MB);   //  2 MB  Wo bf16
  __bf16* qb  = (__bf16*)(ws + 24 * MB);   // 16 MB  Q  [B,H,S,64]
  __bf16* kb  = (__bf16*)(ws + 40 * MB);   // 16 MB  K  [B,H,S,64]
  __bf16* vtb = (__bf16*)(ws + 56 * MB);   // 16 MB  V^T[B,H,64,S]
  __bf16* aob = (__bf16*)(ws + 72 * MB);   // 16 MB  attn out [8192,1024]

  const int nx4 = 8192 * 1024 / 4, nq4 = 3072 * 1024 / 4, no4 = 1024 * 1024 / 4;
  cvt_f32_bf16<<<(nx4 + 255) / 256, 256, 0, stream>>>((const float4*)x, xb, nx4);
  cvt_f32_bf16<<<(nq4 + 255) / 256, 256, 0, stream>>>((const float4*)Wqkv, wqb, nq4);
  cvt_f32_bf16<<<(no4 + 255) / 256, 256, 0, stream>>>((const float4*)Wo_w, wob, no4);

  gemm_qkv<<<dim3(48, 128), 128, 0, stream>>>(xb, wqb, qb, kb, vtb);
  flash_attn<<<dim3(32, 64), 128, 0, stream>>>(qb, kb, vtb, aob);
  gemm_out<<<dim3(16, 128), 128, 0, stream>>>(aob, wob, Wo_b, out);
}
